// EspAttention_3736621547679
// MI455X (gfx1250) — compile-verified, matrix operand order checked by reading, not run on hardware
//
#include <hip/hip_runtime.h>
#include <hip/hip_bf16.h>

// ---------------------------------------------------------------------------
// EspAttention for MI455X (gfx1250, wave32, WMMA).
// B=4, H=8, N=128, DIM_HEAD=L=64, TAU=1e-3, TEMP=10, SCALE=1/8.
// Dominant work: 32*64 batched 128x128x128 matmuls (Gamma = Pu^T Pv) -> f16
// WMMA w/ f32 accumulate. Precision-critical paths (qkv feeding the tau=1e-3
// sort, sort, softmax logits, cost, output) stay f32 scalar.
// ---------------------------------------------------------------------------

typedef __attribute__((ext_vector_type(16))) _Float16 v16h;
typedef __attribute__((ext_vector_type(8)))  _Float16 v8h;
typedef __attribute__((ext_vector_type(8)))  float    v8f;

#define BATCH   4
#define HEADS   8
#define NTOK    128
#define DHEAD   64
#define BH      (BATCH*HEADS)      // 32
#define NCHUNK  8                  // l-chunks per head
#define LPC     (DHEAD/NCHUNK)     // 8 l per chunk

#define SCALEF   0.125f
#define INV_TAU  1000.0f
#define TEMPR    10.0f

// ---- workspace layout (floats) ----
#define OFF_X   0u            // [BH,128,64]  scaled q
#define OFF_Y   262144u       // [BH,128,64]  scaled k
#define OFF_V   524288u       // [BH,128,64]
#define OFF_HO  786432u       // [BH,128,64]  attn @ v
#define OFF_C   1048576u      // [BH,128,128] cost
#define OFF_AP  1572864u      // [BH,NCHUNK,128,128] partial attn accumulators
#define OFF_ZP  5767168u      // [BH*NCHUNK]
#define OFF_RP  5767424u      // [BH*NCHUNK]

// ---- LDS layout for the fused attention kernel (bytes) ----
#define SXP  130              // f32 pitch for transposed X/Y ([l][n])
#define CP   132              // f32 pitch for C and A (bank-spread rows)
#define PP   136              // f16 pitch for PuT/PvT (272B rows, 16B aligned)
#define OFF_SX   0u
#define OFF_SY   33280u
#define OFF_CM   66560u
#define OFF_AM   134144u
#define OFF_PU   201728u
#define OFF_PV   236544u
#define OFF_SU   271360u
#define OFF_SV   271872u
#define OFF_SSU  272384u
#define OFF_SSV  272896u
#define OFF_RED  273408u
#define SMEM_K3  273472u      // < 320KB per WGP

// ===========================================================================
// K1: qkv = x @ w_qkv.T, split/scale into X=q*s, Y=k*s, V  (f32 exact)
// ===========================================================================
__global__ __launch_bounds__(256) void esp_qkv(const float* __restrict__ x,
                                               const float* __restrict__ w_qkv,
                                               float* __restrict__ X,
                                               float* __restrict__ Y,
                                               float* __restrict__ V) {
  __shared__ float xrow[512];
  const int tid = threadIdx.x;
  const int r = blockIdx.x;            // b*128+n
  const int b = r >> 7, n = r & 127;
  for (int t = tid; t < 512; t += 256) xrow[t] = x[r * 512 + t];
  __syncthreads();
  for (int tt = 0; tt < 6; ++tt) {
    const int c = tt * 256 + tid;      // [0,1536)
    const float* wrow = w_qkv + c * 512;
    float acc = 0.f;
    for (int k = 0; k < 512; ++k) acc = fmaf(xrow[k], wrow[k], acc);
    const int part = c >> 9, cc = c & 511, h = cc >> 6, dd = cc & 63;
    const int dst = ((b * HEADS + h) * NTOK + n) * DHEAD + dd;
    if (part == 0)      X[dst] = acc * SCALEF;
    else if (part == 1) Y[dst] = acc * SCALEF;
    else                V[dst] = acc;
  }
}

// ===========================================================================
// K2: per-head cost matrix C[i,j] = ||X_i - Y_j||_2  (f32, computed once)
// ===========================================================================
__global__ __launch_bounds__(256) void esp_cost(const float* __restrict__ X,
                                                const float* __restrict__ Y,
                                                float* __restrict__ Cg) {
  extern __shared__ char smem[];
  float* sXc = (float*)smem;              // [128][64]
  float* sYc = (float*)(smem + 32768);    // [128][64]
  float* nrm = (float*)(smem + 65536);    // [256]
  const int tid = threadIdx.x, bh = blockIdx.x;
  for (int t = tid; t < NTOK * DHEAD; t += 256) {
    sXc[t] = X[bh * NTOK * DHEAD + t];
    sYc[t] = Y[bh * NTOK * DHEAD + t];
  }
  __syncthreads();
  {
    const int i = tid & 127;
    const float* base = (tid < 128) ? sXc : sYc;
    float acc = 0.f;
    for (int dd = 0; dd < DHEAD; ++dd) { float v = base[i * DHEAD + dd]; acc = fmaf(v, v, acc); }
    nrm[tid] = acc;                       // [0..127]=|X_i|^2, [128..255]=|Y_j|^2
  }
  __syncthreads();
  for (int p = 0; p < 64; ++p) {
    const int idx = p * 256 + tid;
    const int i = idx >> 7, j = idx & 127;
    float dot = 0.f;
    for (int dd = 0; dd < DHEAD; ++dd) dot = fmaf(sXc[i * DHEAD + dd], sYc[j * DHEAD + dd], dot);
    const float sq = nrm[i] + nrm[128 + j] - 2.f * dot;
    Cg[bh * NTOK * NTOK + idx] = sqrtf(fmaxf(sq, 0.f));
  }
}

// ===========================================================================
// K3: fused SoftSort + Gamma (f16 WMMA) + online softmin accumulation.
// One block = one (bh, l-chunk). 8 waves; wave w owns the 16-row tile strip.
// ===========================================================================
__global__ __launch_bounds__(256) void esp_attn_chunk(const float* __restrict__ Xg,
                                                      const float* __restrict__ Yg,
                                                      const float* __restrict__ Cg,
                                                      float* __restrict__ Apart,
                                                      float* __restrict__ Zpart,
                                                      float* __restrict__ Rpart) {
  extern __shared__ char smem[];
  float*    sX  = (float*)(smem + OFF_SX);    // [64][SXP] X transposed: [l][n]
  float*    sY  = (float*)(smem + OFF_SY);
  float*    Cm  = (float*)(smem + OFF_CM);    // [128][CP]
  float*    Am  = (float*)(smem + OFF_AM);    // [128][CP] running attn accum
  _Float16* PuT = (_Float16*)(smem + OFF_PU); // [128][PP]  Pu^T : [i][k]
  _Float16* PvT = (_Float16*)(smem + OFF_PV); // [128][PP]  Pv^T : [j][k]
  float*    su  = (float*)(smem + OFF_SU);
  float*    sv  = (float*)(smem + OFF_SV);
  float*    ssu = (float*)(smem + OFF_SSU);
  float*    ssv = (float*)(smem + OFF_SSV);
  float*    red = (float*)(smem + OFF_RED);

  const int tid   = threadIdx.x;
  const int bh    = blockIdx.x / NCHUNK;
  const int chunk = blockIdx.x % NCHUNK;
  const int wave  = tid >> 5;
  const int lane  = tid & 31;

  // load X,Y transposed ([l][n]); padded pitch avoids bank conflicts
  for (int t = tid; t < NTOK * DHEAD; t += 256) {
    const int n = t >> 6, dd = t & 63;
    sX[dd * SXP + n] = Xg[bh * NTOK * DHEAD + t];
    sY[dd * SXP + n] = Yg[bh * NTOK * DHEAD + t];
  }
  for (int t = tid; t < NTOK * NTOK; t += 256) {
    Cm[(t >> 7) * CP + (t & 127)] = Cg[bh * NTOK * NTOK + t];
    Am[(t >> 7) * CP + (t & 127)] = 0.f;
  }
  __syncthreads();

  float rmin = 3.4e38f;   // running min of swds (tracked identically by all threads)
  float Z    = 0.f;       // running softmin normalizer

  const int m0  = wave * 16;
  const int lr  = lane & 15;
  const int klo = (lane < 16) ? 0 : 8;   // 16-bit A/B WMMA VGPR layout K offset
  const int mo8 = (lane < 16) ? 0 : 8;   // f32 C/D layout M offset

  for (int il = 0; il < LPC; ++il) {
    const int l = chunk * LPC + il;

    // ---- stage scores + copies for sorting ----
    if (tid < 128)       { float v = sX[l * SXP + tid];      su[tid] = v;      ssu[tid] = v; }
    else                 { int i = tid - 128;
                           float v = sY[l * SXP + i];        sv[i]   = v;      ssv[i]   = v; }
    __syncthreads();

    // ---- bitonic sort, descending: threads<128 sort ssu, >=128 sort ssv ----
    {
      float* arr = (tid < 128) ? ssu : ssv;
      const int i0 = tid & 127;
      for (int k = 2; k <= 128; k <<= 1) {
        for (int j = k >> 1; j > 0; j >>= 1) {
          const int ixj = i0 ^ j;
          if (ixj > i0) {
            const float a = arr[i0], b = arr[ixj];
            const bool desc = ((i0 & k) == 0);
            const bool sw = desc ? (a < b) : (a > b);
            if (sw) { arr[i0] = b; arr[ixj] = a; }
          }
          __syncthreads();
        }
      }
    }

    // ---- row softmax of exp(-(s_i - sorted_k)^2/tau); max logit is exactly 0
    //      since sorted is a permutation of s. Write TRANSPOSED f16 (WMMA form).
    for (int rr = 0; rr < 16; ++rr) {
      const int kk = wave * 16 + rr;
      const float sku = ssu[kk], skv = ssv[kk];
      float eu[4], ev[4], sumu = 0.f, sumv = 0.f;
#pragma unroll
      for (int t = 0; t < 4; ++t) {
        const int i = lane + 32 * t;
        const float du = su[i] - sku; const float e1 = __expf(-du * du * INV_TAU);
        const float dv = sv[i] - skv; const float e2 = __expf(-dv * dv * INV_TAU);
        eu[t] = e1; sumu += e1; ev[t] = e2; sumv += e2;
      }
#pragma unroll
      for (int off = 16; off > 0; off >>= 1) {
        sumu += __shfl_xor(sumu, off, 32);
        sumv += __shfl_xor(sumv, off, 32);
      }
      const float ru = 1.f / sumu, rv = 1.f / sumv;
#pragma unroll
      for (int t = 0; t < 4; ++t) {
        const int i = lane + 32 * t;
        PuT[i * PP + kk] = (_Float16)(eu[t] * ru);
        PvT[i * PP + kk] = (_Float16)(ev[t] * rv);
      }
    }
    __syncthreads();

    // ---- Gamma = Pu^T Pv via v_wmma_f32_16x16x32_f16; wave w: tile-rows m0 ----
    v16h afr[4];
#pragma unroll
    for (int ks = 0; ks < 4; ++ks) {   // A tile (m0, K=ks*32) from PuT rows
      const _Float16* pa = PuT + (m0 + lr) * PP + ks * 32 + klo;
      const v8h alo = *(const v8h*)pa;
      const v8h ahi = *(const v8h*)(pa + 16);
      afr[ks] = __builtin_shufflevector(alo, ahi, 0,1,2,3,4,5,6,7,8,9,10,11,12,13,14,15);
    }
    float gf[64];
    float swacc = 0.f;
#pragma unroll
    for (int nt = 0; nt < 8; ++nt) {
      v8f c = {0.f,0.f,0.f,0.f,0.f,0.f,0.f,0.f};
#pragma unroll
      for (int ks = 0; ks < 4; ++ks) { // B tile (K=ks*32, nt) from PvT rows
        const _Float16* pb = PvT + (nt * 16 + lr) * PP + ks * 32 + klo;
        const v8h blo = *(const v8h*)pb;
        const v8h bhi = *(const v8h*)(pb + 16);
        const v16h bfr = __builtin_shufflevector(blo, bhi, 0,1,2,3,4,5,6,7,8,9,10,11,12,13,14,15);
        c = __builtin_amdgcn_wmma_f32_16x16x32_f16(false, afr[ks], false, bfr,
                                                   (short)0, c, false, false);
      }
#pragma unroll
      for (int r = 0; r < 8; ++r) {    // D layout: VGPR r -> M = m0+mo8+r, N = nt*16+lr
        const int mi = m0 + mo8 + r;
        const int nj = nt * 16 + lr;
        swacc = fmaf(Cm[mi * CP + nj], c[r], swacc);
        gf[nt * 8 + r] = c[r];
      }
    }

    // ---- block-reduce swds_l ----
#pragma unroll
    for (int off = 16; off > 0; off >>= 1) swacc += __shfl_xor(swacc, off, 32);
    if (lane == 0) red[wave] = swacc;
    __syncthreads();
    float swds = 0.f;
#pragma unroll
    for (int w = 0; w < 8; ++w) swds += red[w];

    // ---- online softmin (flash-style, min instead of max) ----
    float f, e;
    if (swds < rmin) { f = __expf(-TEMPR * (rmin - swds)); e = 1.f; rmin = swds; }
    else             { f = 1.f; e = __expf(-TEMPR * (swds - rmin)); }
    Z = Z * f + e;
#pragma unroll
    for (int nt = 0; nt < 8; ++nt) {
#pragma unroll
      for (int r = 0; r < 8; ++r) {
        const int id = (m0 + mo8 + r) * CP + nt * 16 + lr;
        Am[id] = fmaf(Am[id], f, e * gf[nt * 8 + r]);
      }
    }
    __syncthreads();
  }

  // ---- spill chunk-partials ----
  for (int t = tid; t < NTOK * NTOK; t += 256)
    Apart[(bh * NCHUNK + chunk) * NTOK * NTOK + t] = Am[(t >> 7) * CP + (t & 127)];
  if (tid == 0) {
    Zpart[bh * NCHUNK + chunk] = Z;
    Rpart[bh * NCHUNK + chunk] = rmin;
  }
}

// ===========================================================================
// K4: combine chunks -> attn (written to d_out) and head_out = attn @ V
// ===========================================================================
__global__ __launch_bounds__(256) void esp_combine(const float* __restrict__ Apart,
                                                   const float* __restrict__ Zpart,
                                                   const float* __restrict__ Rpart,
                                                   const float* __restrict__ Vg,
                                                   float* __restrict__ attn_out,
                                                   float* __restrict__ HO) {
  extern __shared__ char smem[];
  float* Am = (float*)smem;             // [128][128] normalized attn
  float* Vl = (float*)(smem + 65536);   // [128][64]
  const int tid = threadIdx.x, bh = blockIdx.x;

  float wc[NCHUNK];
  float r = 3.4e38f;
  for (int c = 0; c < NCHUNK; ++c) r = fminf(r, Rpart[bh * NCHUNK + c]);
  float Zt = 0.f;
  for (int c = 0; c < NCHUNK; ++c) {
    const float w = __expf(-TEMPR * (Rpart[bh * NCHUNK + c] - r));
    wc[c] = w;
    Zt = fmaf(w, Zpart[bh * NCHUNK + c], Zt);
  }
  const float zinv = 1.f / Zt;

  for (int t = tid; t < NTOK * NTOK; t += 256) {
    float acc = 0.f;
    for (int c = 0; c < NCHUNK; ++c)
      acc = fmaf(wc[c], Apart[(bh * NCHUNK + c) * NTOK * NTOK + t], acc);
    const float a = acc * zinv;
    Am[t] = a;
    attn_out[bh * NTOK * NTOK + t] = a;
  }
  for (int t = tid; t < NTOK * DHEAD; t += 256) Vl[t] = Vg[bh * NTOK * DHEAD + t];
  __syncthreads();

  for (int p = 0; p < 32; ++p) {
    const int idx = p * 256 + tid;
    const int n = idx >> 6, dd = idx & 63;
    float acc = 0.f;
    for (int j = 0; j < NTOK; ++j) acc = fmaf(Am[n * NTOK + j], Vl[j * DHEAD + dd], acc);
    HO[bh * NTOK * DHEAD + idx] = acc;
  }
}

// ===========================================================================
// K5: out = head_out @ w_out.T + b_out
// ===========================================================================
__global__ __launch_bounds__(256) void esp_proj(const float* __restrict__ HO,
                                                const float* __restrict__ w_out,
                                                const float* __restrict__ b_out,
                                                float* __restrict__ out) {
  __shared__ float hrow[512];
  const int tid = threadIdx.x;
  const int r = blockIdx.x;             // b*128+n
  const int b = r >> 7, n = r & 127;
  for (int t = tid; t < 512; t += 256) {
    const int h = t >> 6, dd = t & 63;
    hrow[t] = HO[((b * HEADS + h) * NTOK + n) * DHEAD + dd];
  }
  __syncthreads();
  for (int tt = 0; tt < 2; ++tt) {
    const int c = tt * 256 + tid;
    float acc = b_out[c];
    const float* wrow = w_out + c * 512;
    for (int k = 0; k < 512; ++k) acc = fmaf(hrow[k], wrow[k], acc);
    out[r * 512 + c] = acc;
  }
}

// ===========================================================================
extern "C" void kernel_launch(void* const* d_in, const int* in_sizes, int n_in,
                              void* d_out, int out_size, void* d_ws, size_t ws_size,
                              hipStream_t stream) {
  (void)in_sizes; (void)n_in; (void)out_size; (void)ws_size;
  const float* x      = (const float*)d_in[0];
  const float* w_qkv  = (const float*)d_in[1];
  const float* w_out  = (const float*)d_in[2];
  const float* b_out  = (const float*)d_in[3];

  float* out      = (float*)d_out;                       // [4,128,512]
  float* attn_out = out + BATCH * NTOK * 512;            // [4,8,128,128]

  float* ws = (float*)d_ws;                              // needs ~23 MB
  float* X  = ws + OFF_X;
  float* Y  = ws + OFF_Y;
  float* V  = ws + OFF_V;
  float* HO = ws + OFF_HO;
  float* Cg = ws + OFF_C;
  float* AP = ws + OFF_AP;
  float* ZP = ws + OFF_ZP;
  float* RP = ws + OFF_RP;

  esp_qkv       <<<BATCH * NTOK, 256, 0,        stream>>>(x, w_qkv, X, Y, V);
  esp_cost      <<<BH,           256, 66560,    stream>>>(X, Y, Cg);
  esp_attn_chunk<<<BH * NCHUNK,  256, SMEM_K3,  stream>>>(X, Y, Cg, AP, ZP, RP);
  esp_combine   <<<BH,           256, 98304,    stream>>>(AP, ZP, RP, V, attn_out, HO);
  esp_proj      <<<BATCH * NTOK, 256, 0,        stream>>>(HO, w_out, b_out, out);
}
